// DINOVisionMamba_62560493634181
// MI455X (gfx1250) — matrix operand order repaired and checked by reading, not compiled, not run on hardware
//
#include <hip/hip_runtime.h>
#include <hip/hip_bf16.h>

// ---------------------------------------------------------------------------
// Vision Mamba forward for MI455X (gfx1250, wave32, WMMA + TDM).
// GEMMs (>95% of ~580 GFLOP) run on v_wmma_f32_16x16x32_bf16 with f32 acc.
// Big GEMMs stage BOTH A and B panels into LDS with tensor_load_to_lds (TDM),
// double-buffered: steady state is ds_load_b128 -> v_wmma, all global traffic
// on the TDM engine, one s_wait_tensorcnt + one workgroup barrier per K-step.
// ---------------------------------------------------------------------------

typedef __bf16 bf16;
typedef __attribute__((ext_vector_type(16))) __bf16 v16bf;
typedef __attribute__((ext_vector_type(8)))  float  v8f;
typedef __attribute__((ext_vector_type(4)))  unsigned int u32x4;
typedef __attribute__((ext_vector_type(8)))  int  i32x8;
typedef __attribute__((ext_vector_type(4)))  int  i32x4;

#define B_   16
#define D_   768
#define L_   197
#define DI_  1536
#define DS_  16
#define DC_  4
#define DTR_ 48
#define M_   (B_ * L_)        // 3152 token rows
#define NP_  196
#define MPE_ (B_ * NP_)       // 3136 patch rows

union V16U { u32x4 u[2]; v16bf v; };

#if __has_builtin(__builtin_amdgcn_tensor_load_to_lds)
#define USE_TDM 1
#else
#define USE_TDM 0
#endif

#if USE_TDM
// Issue a 2D TDM load: tile (tile_d0 x tile_d1) elements of 2-byte data from
// global (row stride = stride0 elements) into LDS at lds_off. D# per ISA §8.
// OOB rows (beyond tensor_d1) are zero-filled by the TDM.
__device__ __forceinline__ void tdm_load_2d(unsigned lds_off, const void* gaddr,
                                            unsigned tensor_d0, unsigned tensor_d1,
                                            unsigned tile_d0, unsigned tile_d1,
                                            unsigned stride0)
{
    unsigned long long ga = (unsigned long long)(uintptr_t)gaddr;
    u32x4 g0;
    g0[0] = 1u;                                    // count=1, user descriptor
    g0[1] = lds_off;                               // LDS byte address
    g0[2] = (unsigned)(ga & 0xffffffffu);          // global_addr[31:0]
    g0[3] = (unsigned)((ga >> 32) & 0x01ffffffu)   // global_addr[56:32]
            | (2u << 30);                          // type=2 ("image")
    i32x8 g1;
    g1[0] = (int)(1u << 16);                       // data_size=1 -> 2 bytes
    g1[1] = (int)((tensor_d0 & 0xffffu) << 16);    // tensor_dim0[15:0]
    g1[2] = (int)((tensor_d0 >> 16) | ((tensor_d1 & 0xffffu) << 16));
    g1[3] = (int)((tensor_d1 >> 16) | (tile_d0 << 16));   // tile_dim0
    g1[4] = (int)tile_d1;                          // tile_dim1 (tile_dim2=0)
    g1[5] = (int)stride0;                          // tensor_dim0_stride[31:0]
    g1[6] = 0;                                     // stride0 hi / stride1 lo
    g1[7] = 0;
    i32x4 z4 = {0, 0, 0, 0};
#if defined(__clang_major__) && (__clang_major__ >= 23)
    i32x8 z8 = {0, 0, 0, 0, 0, 0, 0, 0};
    __builtin_amdgcn_tensor_load_to_lds(g0, g1, z4, z4, z8, 0);
#else
    __builtin_amdgcn_tensor_load_to_lds(g0, g1, z4, z4, 0);
#endif
}
#endif

// ---------------------------------------------------------------------------
// TDM-staged bf16 WMMA GEMM:  C[M,N] = A[M,K] * W[N,K]^T   (f32 accumulate)
// Requires N % 128 == 0, K % 32 == 0, M % 16 == 0.
// Block: 256 thr = 8 waves, 4(M) x 2(N); wave tile 16x64; block tile 64x128.
// A panel 64x32 (4KB) and B panel 128x32 (8KB), double-buffered via TDM.
// ---------------------------------------------------------------------------
__global__ __launch_bounds__(256)
void gemm_bf16_tdm(const bf16* __restrict__ A, const bf16* __restrict__ W,
                   float* __restrict__ C, int M, int N, int K)
{
    __shared__ bf16 apanel[2][64 * 32];            // 2 x 4 KB
    __shared__ bf16 bpanel[2][128 * 32];           // 2 x 8 KB

    const int lane = threadIdx.x & 31;
    const int wave = threadIdx.x >> 5;
    const int wm = wave & 3;
    const int wn = wave >> 2;
    const int mblk = blockIdx.y * 64;
    const int nblk = blockIdx.x * 128;
    const int m0 = mblk + wm * 16;                 // may exceed M on edge block
    const int half = lane >> 4;
    const int l15 = lane & 15;

    v8f acc0 = {}, acc1 = {}, acc2 = {}, acc3 = {};

    // ---- stage helper: W[nblk.., kk..+32] and A[mblk.., kk..+32] -> LDS ----
#if USE_TDM
#define STAGE(ib, kk)                                                          \
    if (wave == 0) {                                                           \
        tdm_load_2d((unsigned)(uintptr_t)&bpanel[(ib)][0],                     \
                    W + (size_t)nblk * K + (kk), (unsigned)K,                  \
                    (unsigned)(N - nblk), 32u, 128u, (unsigned)K);             \
        tdm_load_2d((unsigned)(uintptr_t)&apanel[(ib)][0],                     \
                    A + (size_t)mblk * K + (kk), (unsigned)K,                  \
                    (unsigned)(M - mblk), 32u, 64u, (unsigned)K);              \
    }
#else
#define STAGE(ib, kk)                                                          \
    {   /* cooperative fallback */                                             \
        for (int t = threadIdx.x; t < 128 * 4; t += 256) {                     \
            int r = t >> 2, cq = t & 3;                                        \
            *(u32x4*)&bpanel[(ib)][r * 32 + cq * 8] =                          \
                *(const u32x4*)(W + (size_t)(nblk + r) * K + (kk) + cq * 8);   \
        }                                                                      \
        for (int t = threadIdx.x; t < 64 * 4; t += 256) {                      \
            int r = t >> 2, cq = t & 3;                                        \
            int gr = (mblk + r < M) ? (mblk + r) : (M - 1);                    \
            *(u32x4*)&apanel[(ib)][r * 32 + cq * 8] =                          \
                *(const u32x4*)(A + (size_t)gr * K + (kk) + cq * 8);           \
        }                                                                      \
    }
#endif

    STAGE(0, 0)

    int ib = 0;
    for (int kk = 0; kk < K; kk += 32, ib ^= 1) {
#if USE_TDM
        if (wave == 0) __builtin_amdgcn_s_wait_tensorcnt(0);
#endif
        __syncthreads();          // panel ib resident; prior reads of ib^1 done
        if (kk + 32 < K) STAGE(ib ^ 1, kk + 32)    // overlap next DMA w/ math

        // A fragment (16x32) from LDS: row wm*16+l15, bytes [32*half, +32)
        const bf16* ap = &apanel[ib][0];
        V16U af;
        af.u[0] = *(const u32x4*)(ap + (wm * 16 + l15) * 32 + 16 * half);
        af.u[1] = *(const u32x4*)(ap + (wm * 16 + l15) * 32 + 16 * half + 8);

        // B fragment j from LDS: row wn*64 + j*16 + l15, bytes [32*half, +32)
        const bf16* bp = &bpanel[ib][0];
        v16bf b0 = *(const v16bf*)(bp + (wn * 64 +  0 + l15) * 32 + 16 * half);
        v16bf b1 = *(const v16bf*)(bp + (wn * 64 + 16 + l15) * 32 + 16 * half);
        v16bf b2 = *(const v16bf*)(bp + (wn * 64 + 32 + l15) * 32 + 16 * half);
        v16bf b3 = *(const v16bf*)(bp + (wn * 64 + 48 + l15) * 32 + 16 * half);

        acc0 = __builtin_amdgcn_wmma_f32_16x16x32_bf16(false, af.v, false, b0, (short)0, acc0, false, false);
        acc1 = __builtin_amdgcn_wmma_f32_16x16x32_bf16(false, af.v, false, b1, (short)0, acc1, false, false);
        acc2 = __builtin_amdgcn_wmma_f32_16x16x32_bf16(false, af.v, false, b2, (short)0, acc2, false, false);
        acc3 = __builtin_amdgcn_wmma_f32_16x16x32_bf16(false, af.v, false, b3, (short)0, acc3, false, false);
    }
#undef STAGE

    // stores: guard edge-block waves (uniform per wave; after all WMMAs)
    if (m0 < M) {
        const int n0 = nblk + wn * 64;
        #define STORE_TILE(J, ACCJ)                                            \
            {                                                                  \
                float* crow = C + (size_t)(m0 + 8 * half) * N                  \
                                + (n0 + (J)*16 + l15);                         \
                _Pragma("unroll")                                              \
                for (int g = 0; g < 8; ++g) crow[(size_t)g * N] = ACCJ[g];     \
            }
        STORE_TILE(0, acc0) STORE_TILE(1, acc1) STORE_TILE(2, acc2) STORE_TILE(3, acc3)
        #undef STORE_TILE
    }
}

// ---------------------------------------------------------------------------
// Direct-load WMMA GEMM (no LDS), wave tile 16 x (16*NT), WM waves in M.
// Used for the small-N x_proj GEMM (N = 80 -> NT=5, WM=8, one N-block).
// ---------------------------------------------------------------------------
template<int NT, int WM>
__global__ __launch_bounds__(256)
void gemm_bf16_direct(const bf16* __restrict__ A, const bf16* __restrict__ W,
                      float* __restrict__ C, int M, int N, int K)
{
    const int lane = threadIdx.x & 31;
    const int wave = threadIdx.x >> 5;
    const int wm = wave % WM;
    const int wn = wave / WM;
    int m0 = (blockIdx.y * WM + wm) * 16;
    if (m0 > M - 16) m0 = M - 16;                  // clamp: EXEC stays full
    const int n0 = (blockIdx.x * (8 / WM) + wn) * (16 * NT);
    const int half = lane >> 4;
    const int mrow = m0 + (lane & 15);
    const int ncol = lane & 15;

    v8f acc[NT];
    #pragma unroll
    for (int j = 0; j < NT; ++j) acc[j] = (v8f){};

    const bf16* arow = A + (size_t)mrow * K;
    for (int kk = 0; kk < K; kk += 32) {
        V16U af;
        af.u[0] = *(const u32x4*)(arow + kk + 16 * half);
        af.u[1] = *(const u32x4*)(arow + kk + 16 * half + 8);
        if (kk + 32 < K) __builtin_prefetch(arow + kk + 32, 0, 1);
        #pragma unroll
        for (int j = 0; j < NT; ++j) {
            const bf16* wrow = W + (size_t)(n0 + j * 16 + ncol) * K + kk + 16 * half;
            v16bf bf = *(const v16bf*)wrow;
            acc[j] = __builtin_amdgcn_wmma_f32_16x16x32_bf16(
                         false, af.v, false, bf, (short)0, acc[j], false, false);
        }
    }
    #pragma unroll
    for (int j = 0; j < NT; ++j) {
        float* crow = C + (size_t)(m0 + 8 * half) * N + (n0 + j * 16 + ncol);
        #pragma unroll
        for (int g = 0; g < 8; ++g) crow[(size_t)g * N] = acc[j][g];
    }
}

// ---------------------------------------------------------------------------
// fp32 -> bf16 conversion
// ---------------------------------------------------------------------------
__global__ __launch_bounds__(256)
void f2bf_kernel(const float* __restrict__ in, bf16* __restrict__ out, int n)
{
    int i = blockIdx.x * 256 + threadIdx.x;
    if (i < n) out[i] = (bf16)in[i];
}

// ---------------------------------------------------------------------------
// dt GEMM K-padding packers (48 -> 64, zero fill) so dt_proj runs on WMMA.
// ---------------------------------------------------------------------------
__global__ __launch_bounds__(256)
void pack_dt_act_kernel(const float* __restrict__ xdbl, bf16* __restrict__ out)
{
    int idx = blockIdx.x * 256 + threadIdx.x;
    if (idx >= M_ * 64) return;
    int r = idx & 63, m = idx >> 6;
    out[idx] = (r < DTR_) ? (bf16)xdbl[(size_t)m * (DTR_ + 2 * DS_) + r] : (bf16)0.0f;
}

__global__ __launch_bounds__(256)
void pack_dt_w_kernel(const float* __restrict__ dw, bf16* __restrict__ out)
{
    int idx = blockIdx.x * 256 + threadIdx.x;
    if (idx >= DI_ * 64) return;
    int r = idx & 63, d = idx >> 6;
    out[idx] = (r < DTR_) ? (bf16)dw[(size_t)d * DTR_ + r] : (bf16)0.0f;
}

// ---------------------------------------------------------------------------
// im2col for patch embed: rows = b*196+p, cols = c*256 + py*16 + px
// ---------------------------------------------------------------------------
__global__ __launch_bounds__(256)
void im2col_kernel(const float* __restrict__ x, bf16* __restrict__ out)
{
    int idx = blockIdx.x * 256 + threadIdx.x;
    if (idx >= MPE_ * D_) return;
    int r = idx / D_, k = idx % D_;
    int b = r / NP_, p = r % NP_;
    int pr = p / 14, pc = p % 14;
    int ch = k / 256, rem = k % 256, py = rem / 16, px = rem % 16;
    float v = x[(((size_t)b * 3 + ch) * 224 + (pr * 16 + py)) * 224 + (pc * 16 + px)];
    out[idx] = (bf16)v;
}

// ---------------------------------------------------------------------------
// Assemble tokens: hidden = [cls; patches] + pos ; resid = 0
// ---------------------------------------------------------------------------
__global__ __launch_bounds__(256)
void assemble_tok_kernel(const float* __restrict__ Cpe, const float* __restrict__ pb,
                         const float* __restrict__ cls, const float* __restrict__ pos,
                         float* __restrict__ hidden, float* __restrict__ resid)
{
    int idx = blockIdx.x * 256 + threadIdx.x;
    if (idx >= M_ * D_) return;
    int d = idx % D_;
    int m = idx / D_;
    int b = m / L_, l = m % L_;
    float v;
    if (l == 0) v = cls[d] + pos[d];
    else        v = Cpe[((size_t)b * NP_ + (l - 1)) * D_ + d] + pb[d]
                    + pos[(size_t)l * D_ + d];
    hidden[idx] = v;
    resid[idx]  = 0.0f;
}

// ---------------------------------------------------------------------------
// Fused: resid += hidden; out_bf = LayerNorm(resid)*w + b   (block per row)
// ---------------------------------------------------------------------------
__global__ __launch_bounds__(256)
void add_ln_kernel(const float* __restrict__ hidden, float* __restrict__ resid,
                   const float* __restrict__ w, const float* __restrict__ b,
                   bf16* __restrict__ out)
{
    __shared__ float rs[8], rs2[8];
    __shared__ float mu_s, rstd_s;
    const size_t base = (size_t)blockIdx.x * D_;
    float v[3], s = 0.f, s2 = 0.f;
    #pragma unroll
    for (int i = 0; i < 3; ++i) {
        int c = threadIdx.x + i * 256;
        float x = resid[base + c] + hidden[base + c];
        v[i] = x; s += x; s2 += x * x;
        resid[base + c] = x;
    }
    for (int off = 16; off > 0; off >>= 1) {
        s  += __shfl_down(s,  off, 32);
        s2 += __shfl_down(s2, off, 32);
    }
    int lane = threadIdx.x & 31, wv = threadIdx.x >> 5;
    if (lane == 0) { rs[wv] = s; rs2[wv] = s2; }
    __syncthreads();
    if (threadIdx.x == 0) {
        float S = 0.f, S2 = 0.f;
        for (int i = 0; i < 8; ++i) { S += rs[i]; S2 += rs2[i]; }
        float mu = S / (float)D_;
        mu_s = mu;
        rstd_s = rsqrtf(S2 / (float)D_ - mu * mu + 1e-5f);
    }
    __syncthreads();
    float mu = mu_s, rstd = rstd_s;
    #pragma unroll
    for (int i = 0; i < 3; ++i) {
        int c = threadIdx.x + i * 256;
        out[base + c] = (bf16)((v[i] - mu) * rstd * w[c] + b[c]);
    }
}

// Final LN: out_f32 = LN(resid + hidden)*w + b
__global__ __launch_bounds__(256)
void final_ln_kernel(const float* __restrict__ hidden, const float* __restrict__ resid,
                     const float* __restrict__ w, const float* __restrict__ b,
                     float* __restrict__ out)
{
    __shared__ float rs[8], rs2[8];
    __shared__ float mu_s, rstd_s;
    const size_t base = (size_t)blockIdx.x * D_;
    float v[3], s = 0.f, s2 = 0.f;
    #pragma unroll
    for (int i = 0; i < 3; ++i) {
        int c = threadIdx.x + i * 256;
        float x = resid[base + c] + hidden[base + c];
        v[i] = x; s += x; s2 += x * x;
    }
    for (int off = 16; off > 0; off >>= 1) {
        s  += __shfl_down(s,  off, 32);
        s2 += __shfl_down(s2, off, 32);
    }
    int lane = threadIdx.x & 31, wv = threadIdx.x >> 5;
    if (lane == 0) { rs[wv] = s; rs2[wv] = s2; }
    __syncthreads();
    if (threadIdx.x == 0) {
        float S = 0.f, S2 = 0.f;
        for (int i = 0; i < 8; ++i) { S += rs[i]; S2 += rs2[i]; }
        float mu = S / (float)D_;
        mu_s = mu;
        rstd_s = rsqrtf(S2 / (float)D_ - mu * mu + 1e-5f);
    }
    __syncthreads();
    float mu = mu_s, rstd = rstd_s;
    #pragma unroll
    for (int i = 0; i < 3; ++i) {
        int c = threadIdx.x + i * 256;
        out[base + c] = (v[i] - mu) * rstd * w[c] + b[c];
    }
}

// ---------------------------------------------------------------------------
// Depthwise causal conv (DC=4) over L + SiLU. xin = xz[:, 0:DI].
// ---------------------------------------------------------------------------
__global__ __launch_bounds__(256)
void conv_silu_kernel(const float* __restrict__ xz, const float* __restrict__ cw,
                      const float* __restrict__ cb, float* __restrict__ xcf,
                      bf16* __restrict__ xcb)
{
    int idx = blockIdx.x * 256 + threadIdx.x;
    if (idx >= M_ * DI_) return;
    int d = idx % DI_;
    int m = idx / DI_;
    int b = m / L_, l = m % L_;
    float acc = cb[d];
    const float* wr = cw + (size_t)d * DC_;
    #pragma unroll
    for (int j = 0; j < DC_; ++j) {
        int li = l - (DC_ - 1) + j;
        if (li >= 0)
            acc += wr[j] * xz[((size_t)(b * L_ + li)) * (2 * DI_) + d];
    }
    float sv = acc * (1.0f / (1.0f + __expf(-acc)));
    xcf[idx] = sv;
    xcb[idx] = (bf16)sv;
}

// ---------------------------------------------------------------------------
// Selective scan: one thread per (b, channel). 197 steps x 16 states.
// dt bias + softplus fused here (dtraw comes from the WMMA dt GEMM).
// ---------------------------------------------------------------------------
__global__ __launch_bounds__(256)
void scan_kernel(const float* __restrict__ dtraw, const float* __restrict__ dtb,
                 const float* __restrict__ xdbl, const float* __restrict__ xcf,
                 const float* __restrict__ A_log, float* __restrict__ ys)
{
    int tid = blockIdx.x * 256 + threadIdx.x;
    if (tid >= B_ * DI_) return;
    int b = tid / DI_, d = tid % DI_;
    float A[DS_], st[DS_];
    #pragma unroll
    for (int s = 0; s < DS_; ++s) {
        A[s] = -__expf(A_log[(size_t)d * DS_ + s]);
        st[s] = 0.0f;
    }
    const float bias = dtb[d];
    for (int l = 0; l < L_; ++l) {
        size_t m = (size_t)b * L_ + l;
        float raw = dtraw[m * DI_ + d] + bias;
        float dt = (raw > 20.0f) ? raw : log1pf(__expf(raw));
        float u  = xcf[m * DI_ + d];
        const float* xr = xdbl + m * (DTR_ + 2 * DS_);
        float y = 0.0f;
        #pragma unroll
        for (int s = 0; s < DS_; ++s) {
            float Bv = xr[DTR_ + s];
            float Cv = xr[DTR_ + DS_ + s];
            st[s] = __expf(dt * A[s]) * st[s] + dt * Bv * u;
            y += st[s] * Cv;
        }
        ys[m * DI_ + d] = y;
    }
}

// ---------------------------------------------------------------------------
// y = (ys + xc * D_skip) * silu(z);  z = xz[:, DI:2*DI]. Output bf16.
// ---------------------------------------------------------------------------
__global__ __launch_bounds__(256)
void gate_kernel(const float* __restrict__ ys, const float* __restrict__ xcf,
                 const float* __restrict__ xz, const float* __restrict__ Dsk,
                 bf16* __restrict__ yb)
{
    int idx = blockIdx.x * 256 + threadIdx.x;
    if (idx >= M_ * DI_) return;
    int d = idx % DI_;
    int m = idx / DI_;
    float z = xz[(size_t)m * (2 * DI_) + DI_ + d];
    float sz = z * (1.0f / (1.0f + __expf(-z)));
    yb[idx] = (bf16)((ys[idx] + xcf[idx] * Dsk[d]) * sz);
}

// ---------------------------------------------------------------------------
// Host orchestration
// ---------------------------------------------------------------------------
static inline unsigned cdiv(unsigned a, unsigned b) { return (a + b - 1) / b; }

extern "C" void kernel_launch(void* const* d_in, const int* in_sizes, int n_in,
                              void* d_out, int out_size, void* d_ws, size_t ws_size,
                              hipStream_t stream)
{
    const float* x        = (const float*)d_in[0];
    const float* patch_w  = (const float*)d_in[1];
    const float* patch_b  = (const float*)d_in[2];
    const float* cls_tok  = (const float*)d_in[3];
    const float* pos_emb  = (const float*)d_in[4];
    const float* norm_w   = (const float*)d_in[5];
    const float* norm_b   = (const float*)d_in[6];
    const float* in_w     = (const float*)d_in[7];
    const float* conv_w   = (const float*)d_in[8];
    const float* conv_b   = (const float*)d_in[9];
    const float* xp_w     = (const float*)d_in[10];
    const float* dtp_w    = (const float*)d_in[11];
    const float* dtp_b    = (const float*)d_in[12];
    const float* A_log    = (const float*)d_in[13];
    const float* D_skip   = (const float*)d_in[14];
    const float* out_w    = (const float*)d_in[15];
    const float* nf_w     = (const float*)d_in[16];
    const float* nf_b     = (const float*)d_in[17];
    (void)in_sizes; (void)n_in; (void)out_size; (void)ws_size;

    char* ws = (char*)d_ws;
    size_t off = 0;
    auto alloc = [&](size_t bytes) -> void* {
        void* p = ws + off;
        off += (bytes + 255) & ~(size_t)255;
        return p;
    };
    float* resid    = (float*)alloc((size_t)M_ * D_ * 4);
    float* hidden   = (float*)alloc((size_t)M_ * D_ * 4);
    bf16*  h_bf     = (bf16*) alloc((size_t)M_ * D_ * 2);
    float* xz       = (float*)alloc((size_t)M_ * 2 * DI_ * 4);
    float* xcf      = (float*)alloc((size_t)M_ * DI_ * 4);
    bf16*  xcb      = (bf16*) alloc((size_t)M_ * DI_ * 2);
    float* xdbl     = (float*)alloc((size_t)M_ * (DTR_ + 2 * DS_) * 4);
    float* dtraw    = (float*)alloc((size_t)M_ * DI_ * 4);
    float* ysb      = (float*)alloc((size_t)M_ * DI_ * 4);
    bf16*  y_bf     = (bf16*) alloc((size_t)M_ * DI_ * 2);
    bf16*  inw_bf   = (bf16*) alloc((size_t)2 * DI_ * D_ * 2);
    bf16*  xpw_bf   = (bf16*) alloc((size_t)(DTR_ + 2 * DS_) * DI_ * 2);
    bf16*  outw_bf  = (bf16*) alloc((size_t)D_ * DI_ * 2);
    bf16*  dtpw_bf  = (bf16*) alloc((size_t)DI_ * 64 * 2);
    bf16*  dtpad_bf = (bf16*) alloc((size_t)M_ * 64 * 2);
    bf16*  pew_bf   = (bf16*) alloc((size_t)D_ * D_ * 2);
    bf16*  acol     = (bf16*) alloc((size_t)MPE_ * D_ * 2);
    float* cpe      = (float*)alloc((size_t)MPE_ * D_ * 4);

    const dim3 blk(256);

    // ---- patch embed (N=768 %128==0 -> TDM GEMM) ----
    f2bf_kernel<<<cdiv(D_ * D_, 256), blk, 0, stream>>>(patch_w, pew_bf, D_ * D_);
    im2col_kernel<<<cdiv(MPE_ * D_, 256), blk, 0, stream>>>(x, acol);
    {
        dim3 grid(D_ / 128, cdiv(MPE_, 64));
        gemm_bf16_tdm<<<grid, blk, 0, stream>>>(acol, pew_bf, cpe, MPE_, D_, D_);
    }
    assemble_tok_kernel<<<cdiv(M_ * D_, 256), blk, 0, stream>>>(
        cpe, patch_b, cls_tok, pos_emb, hidden, resid);

    // ---- 24 Mamba blocks ----
    for (int layer = 0; layer < 24; ++layer) {
        const float* lw_in  = in_w  + (size_t)layer * 2 * DI_ * D_;
        const float* lw_xp  = xp_w  + (size_t)layer * (DTR_ + 2 * DS_) * DI_;
        const float* lw_out = out_w + (size_t)layer * D_ * DI_;

        f2bf_kernel<<<cdiv(2 * DI_ * D_, 256), blk, 0, stream>>>(lw_in, inw_bf, 2 * DI_ * D_);
        f2bf_kernel<<<cdiv((DTR_ + 2 * DS_) * DI_, 256), blk, 0, stream>>>(lw_xp, xpw_bf, (DTR_ + 2 * DS_) * DI_);
        f2bf_kernel<<<cdiv(D_ * DI_, 256), blk, 0, stream>>>(lw_out, outw_bf, D_ * DI_);
        pack_dt_w_kernel<<<cdiv(DI_ * 64, 256), blk, 0, stream>>>(
            dtp_w + (size_t)layer * DI_ * DTR_, dtpw_bf);

        add_ln_kernel<<<M_, blk, 0, stream>>>(
            hidden, resid, norm_w + (size_t)layer * D_, norm_b + (size_t)layer * D_, h_bf);

        // xz = h @ in_w^T   (3152 x 3072 x 768) -> TDM GEMM
        {
            dim3 grid((2 * DI_) / 128, cdiv(M_, 64));
            gemm_bf16_tdm<<<grid, blk, 0, stream>>>(h_bf, inw_bf, xz, M_, 2 * DI_, D_);
        }
        conv_silu_kernel<<<cdiv(M_ * DI_, 256), blk, 0, stream>>>(
            xz, conv_w + (size_t)layer * DI_ * DC_, conv_b + (size_t)layer * DI_, xcf, xcb);
        // xdbl = xc @ xp_w^T  (3152 x 80 x 1536): direct, NT=5, WM=8
        {
            dim3 grid(1, cdiv(M_, 128));
            gemm_bf16_direct<5, 8><<<grid, blk, 0, stream>>>(
                xcb, xpw_bf, xdbl, M_, DTR_ + 2 * DS_, DI_);
        }
        // dtraw = dt_pad @ dtp_w_pad^T  (3152 x 1536 x 64) -> TDM GEMM (WMMA)
        pack_dt_act_kernel<<<cdiv(M_ * 64, 256), blk, 0, stream>>>(xdbl, dtpad_bf);
        {
            dim3 grid(DI_ / 128, cdiv(M_, 64));
            gemm_bf16_tdm<<<grid, blk, 0, stream>>>(dtpad_bf, dtpw_bf, dtraw, M_, DI_, 64);
        }
        // selective scan (bias + softplus fused)
        scan_kernel<<<cdiv(B_ * DI_, 256), blk, 0, stream>>>(
            dtraw, dtp_b + (size_t)layer * DI_, xdbl, xcf,
            A_log + (size_t)layer * DI_ * DS_, ysb);
        gate_kernel<<<cdiv(M_ * DI_, 256), blk, 0, stream>>>(
            ysb, xcf, xz, D_skip + (size_t)layer * DI_, y_bf);
        // hidden = y @ out_w^T  (3152 x 768 x 1536) -> TDM GEMM
        {
            dim3 grid(D_ / 128, cdiv(M_, 64));
            gemm_bf16_tdm<<<grid, blk, 0, stream>>>(y_bf, outw_bf, hidden, M_, D_, DI_);
        }
    }

    final_ln_kernel<<<M_, blk, 0, stream>>>(hidden, resid, nf_w, nf_b, (float*)d_out);
}
